// cross_scale_trans_16827681866282
// MI455X (gfx1250) — matrix operand order repaired and checked by reading, not compiled, not run on hardware
//
#include <hip/hip_runtime.h>
#include <math.h>

typedef __bf16 bf16;
typedef __attribute__((ext_vector_type(16))) __bf16 v16bf;
typedef __attribute__((ext_vector_type(8)))  __bf16 v8bf;
typedef __attribute__((ext_vector_type(8)))  float  v8f;

#define NVOX   120000
#define SD     21
#define SH     400
#define SW     352
#define DHW    (SD*SH*SW)

// ---------------------------------------------------------------- utilities
__global__ void k_fill_idx(int* __restrict__ m, int n) {
    int i = blockIdx.x * 256 + threadIdx.x;
    if (i < n) m[i] = -1;
}

__global__ void k_scatter_idx(const int* __restrict__ idx, int* __restrict__ m, int n) {
    int i = blockIdx.x * 256 + threadIdx.x;
    if (i < n) {
        int d = idx[i*3 + 0], h = idx[i*3 + 1], w = idx[i*3 + 2];
        m[d*(SH*SW) + h*SW + w] = i;
    }
}

__global__ void k_cvt_bf16(const float* __restrict__ in, bf16* __restrict__ out, int n) {
    int i = blockIdx.x * 256 + threadIdx.x;
    if (i < n) out[i] = (bf16)in[i];
}

// group_norm with group size 2 over 128 channels -> bf16 output
__global__ void k_groupnorm_pair(const float* __restrict__ in,
                                 const float* __restrict__ g, const float* __restrict__ b,
                                 bf16* __restrict__ out, int n) {
    int i = blockIdx.x * 256 + threadIdx.x;          // one thread per (row, pair)
    if (i >= n * 64) return;
    int row = i >> 6, gp = i & 63;
    int base = row * 128 + gp * 2;
    float x0 = in[base], x1 = in[base + 1];
    float m  = 0.5f * (x0 + x1);
    float d0 = x0 - m, d1 = x1 - m;
    float s  = rsqrtf(0.5f * (d0*d0 + d1*d1) + 1e-5f);
    out[base]     = (bf16)(d0 * s * g[gp*2]     + b[gp*2]);
    out[base + 1] = (bf16)(d1 * s * g[gp*2 + 1] + b[gp*2 + 1]);
}

// ---------------------------------------------------------------- WMMA GEMM
// out[N, n0+n1] = A[N,K](bf16) @ concat(B0[K,n0], B1[K,n1])(f32->bf16) + bias (+resid)(opt relu)
// workgroup: 8 waves x 16 rows = 128 rows; each wave: NTILE column tiles (NTILE*16 cols).
#define APAD 72   // bf16 stride (144 B): 16-B aligned, conflict-free across 16 rows
#define BPAD 72

template <int NTILE>
__global__ __launch_bounds__(256)
void k_gemm_bf16(const bf16* __restrict__ A, int lda, int K,
                 const float* __restrict__ B0, int n0,
                 const float* __restrict__ B1, int n1,
                 const float* __restrict__ bias0, const float* __restrict__ bias1,
                 const float* __restrict__ resid, int ldr,
                 int do_relu,
                 float* __restrict__ outf, bf16* __restrict__ outb,
                 int nrows)
{
    constexpr int COLS = NTILE * 16;
    constexpr int CSH  = (COLS == 128) ? 7 : 6;
    __shared__ bf16 sA [128 * APAD];       // row-major rows x 64-deep K chunk
    __shared__ bf16 sBt[COLS * BPAD];      // TRANSPOSED: [col][k]

    const int ncols = n0 + n1;
    const int row0  = blockIdx.x * 128;
    const int c0    = blockIdx.y * COLS;
    const int tid   = threadIdx.x;
    const int wave  = tid >> 5;
    const int lane  = tid & 31;
    const int hf    = lane >> 4;
    const int l15   = lane & 15;

    v8f acc[NTILE] = {};

    for (int kk = 0; kk < K; kk += 64) {
        // stage A chunk: 8 bf16 (16 B) per thread per step -> b128 load + b128 LDS store
        #pragma unroll
        for (int e4 = tid; e4 < 128 * 8; e4 += 256) {
            int r = e4 >> 3, c = (e4 & 7) * 8;
            int row = row0 + r;
            v8bf av = {};
            if (row < nrows)
                av = *reinterpret_cast<const v8bf*>(&A[(long)row * lda + kk + c]);
            *reinterpret_cast<v8bf*>(&sA[r * APAD + c]) = av;
        }
        // stage B chunk transposed: float4 global read, 4 b16 transposed LDS stores
        // (n0 is always a multiple of 4, rows are 16-B aligned)
        #pragma unroll
        for (int e = tid; e < COLS * 16; e += 256) {
            int gi = e * 4;
            int r = gi >> CSH, c = gi & (COLS - 1);
            int gc = c0 + c;
            const float* src = (gc < n0) ? &B0[(long)(kk + r) * n0 + gc]
                                         : &B1[(long)(kk + r) * n1 + (gc - n0)];
            float4 v4 = *reinterpret_cast<const float4*>(src);
            sBt[(c + 0) * BPAD + r] = (bf16)v4.x;
            sBt[(c + 1) * BPAD + r] = (bf16)v4.y;
            sBt[(c + 2) * BPAD + r] = (bf16)v4.z;
            sBt[(c + 3) * BPAD + r] = (bf16)v4.w;
        }
        __syncthreads();

        // prefetch next K-chunk of A into cache while this chunk computes
        if (kk + 64 < K && tid < 128) {
            int row = row0 + tid;
            if (row < nrows)
                __builtin_prefetch(&A[(long)row * lda + kk + 64], 0, 0);
        }

        #pragma unroll
        for (int s = 0; s < 2; ++s) {
            const int k2 = s * 32;
            // A fragment: element j -> k = k2 + hf*8 + j (j<8), +16 for high half
            const int abase = (wave * 16 + l15) * APAD + k2 + hf * 8;
            v8bf a_lo = *reinterpret_cast<const v8bf*>(&sA[abase]);
            v8bf a_hi = *reinterpret_cast<const v8bf*>(&sA[abase + 16]);
            v16bf afrag = __builtin_shufflevector(a_lo, a_hi,
                0,1,2,3,4,5,6,7,8,9,10,11,12,13,14,15);
            // B fragments double-buffered: issue tile t+1 loads before WMMA t
            const int bb0 = l15 * BPAD + k2 + hf * 8;
            v8bf b_lo = *reinterpret_cast<const v8bf*>(&sBt[bb0]);
            v8bf b_hi = *reinterpret_cast<const v8bf*>(&sBt[bb0 + 16]);
            #pragma unroll
            for (int t = 0; t < NTILE; ++t) {
                v16bf bfrag = __builtin_shufflevector(b_lo, b_hi,
                    0,1,2,3,4,5,6,7,8,9,10,11,12,13,14,15);
                if (t + 1 < NTILE) {
                    const int bbn = ((t + 1) * 16 + l15) * BPAD + k2 + hf * 8;
                    b_lo = *reinterpret_cast<const v8bf*>(&sBt[bbn]);
                    b_hi = *reinterpret_cast<const v8bf*>(&sBt[bbn + 16]);
                }
                acc[t] = __builtin_amdgcn_wmma_f32_16x16x32_bf16(
                    false, afrag, false, bfrag, (short)0, acc[t], false, false);
            }
        }
        __syncthreads();
    }

    // epilogue: C/D element r -> M = r + 8*half, N = lane&15
    #pragma unroll
    for (int t = 0; t < NTILE; ++t) {
        #pragma unroll
        for (int r = 0; r < 8; ++r) {
            int m   = r + hf * 8;
            int row = row0 + wave * 16 + m;
            int col = c0 + t * 16 + l15;
            if (row < nrows) {
                float v = acc[t][r];
                v += (col < n0) ? bias0[col] : bias1[col - n0];
                if (resid)   v += resid[(long)row * ldr + col];
                if (do_relu) v = fmaxf(v, 0.f);
                if (outf) outf[(long)row * ncols + col] = v;
                if (outb) outb[(long)row * ncols + col] = (bf16)v;
            }
        }
    }
}

// --------------------------------------------- softmax + pixel/bounds + index lookup
__global__ void k_prep_sample(const float* __restrict__ offatt,
                              const int* __restrict__ indices,
                              const int* __restrict__ idxmap,
                              int* __restrict__ rid, float* __restrict__ wt, int n) {
    int i = blockIdx.x * 256 + threadIdx.x;
    if (i >= n * 4) return;
    int row = i >> 2, h = i & 3;
    const float* oa = offatt + (long)row * 64;

    float a[4];
    float mx = -1e30f;
    #pragma unroll
    for (int p = 0; p < 4; ++p) { a[p] = oa[48 + h * 4 + p]; mx = fmaxf(mx, a[p]); }
    float sum = 0.f;
    #pragma unroll
    for (int p = 0; p < 4; ++p) { a[p] = expf(a[p] - mx); sum += a[p]; }
    float rs = 1.f / sum;

    float fd = (float)indices[row*3 + 0];
    float fh = (float)indices[row*3 + 1];
    float fw = (float)indices[row*3 + 2];

    #pragma unroll
    for (int p = 0; p < 4; ++p) {
        const float* o = oa + h * 12 + p * 3;          // (h, p, 3) packing of 48 cols
        float pd = rintf(o[0] + fd);
        float ph = rintf(o[1] + fh);
        float pw = rintf(o[2] + fw);
        bool inb = (pd >= 0.f) & (pd <= (float)(SD-1)) &
                   (ph >= 0.f) & (ph <= (float)(SH-1)) &
                   (pw >= 0.f) & (pw <= (float)(SW-1));
        int cd = (int)fminf(fmaxf(pd, 0.f), (float)(SD-1));
        int ch = (int)fminf(fmaxf(ph, 0.f), (float)(SH-1));
        int cw = (int)fminf(fmaxf(pw, 0.f), (float)(SW-1));
        int r2 = idxmap[cd * (SH*SW) + ch * SW + cw];
        float w = a[p] * rs * (inb ? 1.f : 0.f);
        if (r2 < 0) w = 0.f;                           // empty dense cell == zero value
        rid[row*16 + h*4 + p] = r2;
        wt [row*16 + h*4 + p] = w;
    }
}

// ---------------------------------------------- wave-per-voxel weighted value gather
__global__ __launch_bounds__(256)
void k_gather_agg(const int* __restrict__ rid, const float* __restrict__ wt,
                  const float* __restrict__ qv, bf16* __restrict__ agg, int n) {
    int wv   = (blockIdx.x * 256 + threadIdx.x) >> 5;
    int lane = threadIdx.x & 31;
    if (wv >= n) return;
    float a0 = 0.f, a1 = 0.f;
    const int*   rp = rid + (long)wv * 16;
    const float* wp = wt  + (long)wv * 16;
    for (int p = 0; p < 16; ++p) {
        int r = rp[p]; float w = wp[p];                // uniform across the wave
        if (r >= 0 && w != 0.f) {
            const float* vrow = qv + (long)r * 128 + 64;   // value = cols 64..127 of qv
            a0 += w * vrow[lane];
            a1 += w * vrow[32 + lane];
        }
    }
    agg[(long)wv * 64 + lane]      = (bf16)a0;
    agg[(long)wv * 64 + 32 + lane] = (bf16)a1;
}

// --------------------------------------------------- wave-per-row LayerNorm (64 ch)
__global__ __launch_bounds__(256)
void k_layernorm(const float* __restrict__ in,
                 const float* __restrict__ g, const float* __restrict__ b,
                 float* __restrict__ outf, bf16* __restrict__ outb, int n) {
    int wv   = (blockIdx.x * 256 + threadIdx.x) >> 5;
    int lane = threadIdx.x & 31;
    if (wv >= n) return;
    float x0 = in[(long)wv * 64 + lane];
    float x1 = in[(long)wv * 64 + 32 + lane];
    float s = x0 + x1;
    #pragma unroll
    for (int o = 16; o > 0; o >>= 1) s += __shfl_xor(s, o, 32);
    float mean = s * (1.f / 64.f);
    float d0 = x0 - mean, d1 = x1 - mean;
    float vs = d0*d0 + d1*d1;
    #pragma unroll
    for (int o = 16; o > 0; o >>= 1) vs += __shfl_xor(vs, o, 32);
    float inv = rsqrtf(vs * (1.f / 64.f) + 1e-5f);
    float y0 = d0 * inv * g[lane]      + b[lane];
    float y1 = d1 * inv * g[32 + lane] + b[32 + lane];
    if (outf) { outf[(long)wv*64 + lane] = y0; outf[(long)wv*64 + 32 + lane] = y1; }
    if (outb) { outb[(long)wv*64 + lane] = (bf16)y0; outb[(long)wv*64 + 32 + lane] = (bf16)y1; }
}

// ------------------------------------------------------------------- launcher
extern "C" void kernel_launch(void* const* d_in, const int* in_sizes, int n_in,
                              void* d_out, int out_size, void* d_ws, size_t ws_size,
                              hipStream_t stream) {
    const float* features = (const float*)d_in[0];
    const int*   indices  = (const int*)  d_in[1];
    const float* W_in  = (const float*)d_in[2];
    const float* b_in  = (const float*)d_in[3];
    const float* gn_g  = (const float*)d_in[4];
    const float* gn_b  = (const float*)d_in[5];
    const float* W_q   = (const float*)d_in[6];
    const float* b_q   = (const float*)d_in[7];
    const float* W_v   = (const float*)d_in[8];
    const float* b_v   = (const float*)d_in[9];
    const float* W_off = (const float*)d_in[10];
    const float* b_off = (const float*)d_in[11];
    const float* W_att = (const float*)d_in[12];
    const float* b_att = (const float*)d_in[13];
    const float* W_out = (const float*)d_in[14];
    const float* b_out = (const float*)d_in[15];
    const float* ln1_g = (const float*)d_in[16];
    const float* ln1_b = (const float*)d_in[17];
    const float* W_f1  = (const float*)d_in[18];
    const float* b_f1  = (const float*)d_in[19];
    const float* W_f2  = (const float*)d_in[20];
    const float* b_f2  = (const float*)d_in[21];
    const float* ln2_g = (const float*)d_in[22];
    const float* ln2_b = (const float*)d_in[23];

    const int N = NVOX;
    char* ws = (char*)d_ws;
    // scratch layout (lifetimes reused; ~258 MB total)
    int*   idxmap  = (int*)  (ws + 0);                       // 11.8 MB
    char*  bufA    =          ws + 11827456;                 // 15.36 MB: feat_bf -> rid/wt
    bf16*  feat_bf = (bf16*) bufA;
    int*   rid     = (int*)   bufA;                          // reuse after GEMM1
    float* wt      = (float*)(bufA + 7680000);
    char*  bufB    =          ws + 27187456;                 // 30.72 MB: src_bf -> agg_bf
    bf16*  src_bf  = (bf16*) bufB;
    bf16*  agg_bf  = (bf16*) bufB;                           // reuse after GEMM2
    float* qv      = (float*)(ws + 57907456);                // 61.44 MB: q|v f32 -> pre2
    float* pre2    = qv;                                     // reuse after GEMM4
    bf16*  qv_bf   = (bf16*) (ws + 119347456);               // 30.72 MB
    char*  bufC    =          ws + 150067456;                // 61.44 MB rotating scratch
    float* src_pre = (float*)bufC;
    float* offatt  = (float*)bufC;
    float* pre1    = (float*)bufC;
    bf16*  h_bf    = (bf16*) bufC;
    float* tgt     = (float*)(ws + 211507456);               // 30.72 MB
    bf16*  tgt_bf  = (bf16*) (ws + 242227456);               // 15.36 MB

    const dim3 blk(256);
    const int rowBlocks = (N + 127) / 128;                   // 938

    // sparse index map: (d,h,w) -> voxel row (replaces 757 MB dense tensor)
    k_fill_idx   <<<(DHW + 255)/256, blk, 0, stream>>>(idxmap, DHW);
    k_scatter_idx<<<(N   + 255)/256, blk, 0, stream>>>(indices, idxmap, N);

    // features -> bf16
    k_cvt_bf16<<<(N*64 + 255)/256, blk, 0, stream>>>(features, feat_bf, N*64);

    // src_pre = features @ W_in + b_in
    k_gemm_bf16<8><<<dim3(rowBlocks,1), blk, 0, stream>>>(feat_bf, 64, 64,
        W_in, 128, nullptr, 0, b_in, nullptr, nullptr, 0, 0, src_pre, nullptr, N);
    // src = group_norm(src_pre) (group size 2), bf16
    k_groupnorm_pair<<<(N*64 + 255)/256, blk, 0, stream>>>(src_pre, gn_g, gn_b, src_bf, N);

    // qv = src @ [W_q | W_v] + [b_q | b_v]   (f32 + bf16 copies)
    k_gemm_bf16<8><<<dim3(rowBlocks,1), blk, 0, stream>>>(src_bf, 128, 128,
        W_q, 64, W_v, 64, b_q, b_v, nullptr, 0, 0, qv, qv_bf, N);

    // offatt = query @ [W_off | W_att] + bias
    k_gemm_bf16<4><<<dim3(rowBlocks,1), blk, 0, stream>>>(qv_bf, 128, 64,
        W_off, 48, W_att, 16, b_off, b_att, nullptr, 0, 0, offatt, nullptr, N);

    // softmax + pixel rounding + bounds + sparse lookup
    k_prep_sample<<<(N*4 + 255)/256, blk, 0, stream>>>(offatt, indices, idxmap, rid, wt, N);

    // agg[n,c] = sum_p wt * value[rid, c]
    k_gather_agg<<<(N + 7)/8, blk, 0, stream>>>(rid, wt, qv, agg_bf, N);

    // pre1 = query + agg @ W_out + b_out
    k_gemm_bf16<4><<<dim3(rowBlocks,1), blk, 0, stream>>>(agg_bf, 64, 64,
        W_out, 64, nullptr, 0, b_out, nullptr, qv, 128, 0, pre1, nullptr, N);

    // tgt = LN1(pre1)
    k_layernorm<<<(N + 7)/8, blk, 0, stream>>>(pre1, ln1_g, ln1_b, tgt, tgt_bf, N);

    // h = relu(tgt @ W_f1 + b_f1)  (bf16 only)
    k_gemm_bf16<8><<<dim3(rowBlocks,2), blk, 0, stream>>>(tgt_bf, 64, 64,
        W_f1, 256, nullptr, 0, b_f1, nullptr, nullptr, 0, 1, nullptr, h_bf, N);

    // pre2 = tgt + h @ W_f2 + b_f2
    k_gemm_bf16<4><<<dim3(rowBlocks,1), blk, 0, stream>>>(h_bf, 256, 256,
        W_f2, 64, nullptr, 0, b_f2, nullptr, tgt, 64, 0, pre2, nullptr, N);

    // out = LN2(pre2)
    k_layernorm<<<(N + 7)/8, blk, 0, stream>>>(pre2, ln2_g, ln2_b, (float*)d_out, nullptr, N);
}
